// Llama4VisionAttention_75290776699001
// MI455X (gfx1250) — matrix operand order repaired and checked by reading, not compiled, not run on hardware
//
#include <hip/hip_runtime.h>
#include <hip/hip_bf16.h>

#define DEVI __device__ __forceinline__

typedef __attribute__((ext_vector_type(16))) __bf16       v16bf;
typedef __attribute__((ext_vector_type(8)))  float        v8f;
typedef __attribute__((ext_vector_type(8)))  unsigned int u32x8;

constexpr int E_c  = 1408;
constexpr int H_c  = 16;
constexpr int D_c  = 88;
constexpr int S_c  = 1025;
constexpr int B_c  = 8;
constexpr int BS_c = B_c * S_c;   // 8200 tokens
constexpr int Dp_c = 96;          // head dim padded to 3*32
constexpr int Sp_c = 1056;        // seq padded to 33*32
constexpr int KSTEP = 32;
constexpr int TDS  = 36;          // LDS row stride (ushorts) incl. TDM pad (64B+8B)

#define HAS_TDM (__has_builtin(__builtin_amdgcn_tensor_load_to_lds) && \
                 __has_builtin(__builtin_amdgcn_s_wait_tensorcnt))

DEVI unsigned short f2bf(float f) {
  union { float f; unsigned int u; } v; v.f = f;
  unsigned int u = v.u;
  u += 0x7FFFu + ((u >> 16) & 1u);   // round-to-nearest-even
  return (unsigned short)(u >> 16);
}

DEVI v8f wmma_bf16(v16bf a, v16bf b, v8f c) {
  // D = A(16x32 bf16) * B(32x16 bf16) + C(16x16 f32)
  return __builtin_amdgcn_wmma_f32_16x16x32_bf16(false, a, false, b, (short)0, c,
                                                 false, false);
}

// ---- WMMA fragment loaders (contiguous k = 16*half + s, same in A and B) ----
// A fragment: 16x32 bf16 tile, source row-major with row stride ld (elements).
DEVI v16bf frag_a(const unsigned short* p, int ld, int lane) {
  const int m = lane & 15, half = lane >> 4;
  u32x8 t;
#pragma unroll
  for (int v = 0; v < 8; ++v) {
    const int k = (half << 4) + (v << 1);
    t[v] = *(const unsigned int*)(p + (size_t)m * ld + k);
  }
  return __builtin_bit_cast(v16bf, t);
}
// B fragment: 32x16 (KxN), source stored [N][K] (K pairs contiguous).
DEVI v16bf frag_b_nk(const unsigned short* p, int ld, int lane) {
  const int n = lane & 15, kb = (lane >> 4) << 4;
  u32x8 t;
#pragma unroll
  for (int v = 0; v < 8; ++v)
    t[v] = *(const unsigned int*)(p + (size_t)n * ld + kb + (v << 1));
  return __builtin_bit_cast(v16bf, t);
}
// B fragment: 32x16 (KxN), source stored [K][N] (pack K pairs manually).
DEVI v16bf frag_b_kn(const unsigned short* p, int ld, int lane) {
  const int n = lane & 15, kb = (lane >> 4) << 4;
  u32x8 t;
#pragma unroll
  for (int v = 0; v < 8; ++v) {
    unsigned int lo = p[(size_t)(kb + (v << 1)) * ld + n];
    unsigned int hi = p[(size_t)(kb + (v << 1) + 1) * ld + n];
    t[v] = lo | (hi << 16);
  }
  return __builtin_bit_cast(v16bf, t);
}

// ---- Tensor Data Mover: build D# per ISA 08_async_tensor 8.3-8.5 and issue ----
#if HAS_TDM
typedef __attribute__((ext_vector_type(4))) unsigned int tdm_u32x4;
typedef __attribute__((ext_vector_type(4))) int          tdm_i32x4;
typedef __attribute__((ext_vector_type(8))) int          tdm_i32x8;

// 2D tile load: tile0 x tile1 elements (2-byte), tensor dims for OOB zero-fill,
// strideElems = row stride in elements. padRows inserts 2 DWORDs per 16 DWORDs
// (i.e. per 64B row) so the LDS row stride becomes TDS=36 ushorts.
DEVI void tdm_load_2d(unsigned int ldsAddr, const void* gptr,
                      unsigned int tdim0, unsigned int tdim1,
                      unsigned int tile0, unsigned int tile1,
                      unsigned long long strideElems, bool padRows) {
  const unsigned long long ga = (unsigned long long)(size_t)gptr;
  tdm_u32x4 g0;
  g0[0] = 1u;                                              // count=1, user D#
  g0[1] = ldsAddr;                                         // lds_addr
  g0[2] = (unsigned int)ga;                                // global_addr[31:0]
  g0[3] = (unsigned int)((ga >> 32) & 0x01FFFFFFu) | (2u << 30);  // addr[56:32]|type=2
  unsigned int f = (1u << 16);                             // data_size=1 -> 2 bytes
  if (padRows) f |= (1u << 20) | (3u << 22) | (1u << 25);  // pad_en, int=16DW, amt=2DW
  tdm_i32x8 g1;
  g1[0] = (int)f;
  g1[1] = (int)((tdim0 & 0xFFFFu) << 16);                  // tensor_dim0[15:0]
  g1[2] = (int)((tdim0 >> 16) | ((tdim1 & 0xFFFFu) << 16));
  g1[3] = (int)((tdim1 >> 16) | (tile0 << 16));            // tile_dim0
  g1[4] = (int)(tile1 & 0xFFFFu);                          // tile_dim1 (tile_dim2=0)
  g1[5] = (int)(unsigned int)(strideElems & 0xFFFFFFFFull);// dim0_stride[31:0]
  g1[6] = (int)(unsigned int)((strideElems >> 32) & 0xFFFFull);
  g1[7] = 0;                                               // dim1_stride unused (2D)
  tdm_i32x4 gz; gz[0] = gz[1] = gz[2] = gz[3] = 0;
#if __clang_major__ >= 23
  tdm_i32x8 gz8;
#pragma unroll
  for (int i = 0; i < 8; ++i) gz8[i] = 0;
  __builtin_amdgcn_tensor_load_to_lds(g0, g1, gz, gz, gz8, 0);
#else
  __builtin_amdgcn_tensor_load_to_lds(g0, g1, gz, gz, 0);
#endif
}
#endif  // HAS_TDM

__global__ void cvt_f32_bf16(const float* __restrict__ src,
                             unsigned short* __restrict__ dst, long n) {
  long i = (long)blockIdx.x * blockDim.x + threadIdx.x;
  const long stride = (long)gridDim.x * blockDim.x;
  for (; i < n; i += stride) dst[i] = f2bf(src[i]);
}

__global__ void zero_u16(unsigned short* __restrict__ dst, long n) {
  long i = (long)blockIdx.x * blockDim.x + threadIdx.x;
  const long stride = (long)gridDim.x * blockDim.x;
  for (; i < n; i += stride) dst[i] = 0;
}

enum { EPI_ROPE = 0, EPI_V = 1, EPI_OUT = 2 };

// Y = A(M x E) * W^T(E x E) + bias ; templated epilogue.
// Block tile 128x128, 8 waves, each wave 64(M) x 32(N) = 4x2 WMMA subtiles.
// TDM double-buffers the 128x32 A/B tiles; DMA of tile kt+1 overlaps WMMA of kt.
template <int EPI>
__global__ __launch_bounds__(256) void gemm_kernel(
    const unsigned short* __restrict__ A,   // [M x E] bf16
    const unsigned short* __restrict__ W,   // [E x E] bf16 row-major [n][k]
    const float* __restrict__ bias,         // [E]
    const float* __restrict__ fcos,         // [S x D/2] (rope only)
    const float* __restrict__ fsin,
    unsigned short* __restrict__ dstBf,     // rope/V dest: [B,H,Sp,Dp] bf16
    float* __restrict__ dstF)               // O dest: [M x E] f32
{
  __shared__ unsigned short As[2][128][TDS];
  __shared__ unsigned short Bs[2][128][TDS];

  const int lane  = threadIdx.x & 31;
  const int wv    = threadIdx.x >> 5;   // 0..7
  const int wm    = wv & 1;             // which 64-row half
  const int wn    = wv >> 1;            // which 32-col quarter
  const int mBase = blockIdx.y * 128;
  const int nBase = blockIdx.x * 128;
  constexpr int NK = E_c / KSTEP;       // 44

  v8f zero8 = {0.f, 0.f, 0.f, 0.f, 0.f, 0.f, 0.f, 0.f};
  v8f acc[4][2];
#pragma unroll
  for (int i = 0; i < 4; ++i)
#pragma unroll
    for (int j = 0; j < 2; ++j) acc[i][j] = zero8;

#if HAS_TDM
  const unsigned int remRows = (unsigned int)(BS_c - mBase < 128 ? BS_c - mBase : 128);
  unsigned int ldsA[2], ldsB[2];
#pragma unroll
  for (int bfi = 0; bfi < 2; ++bfi) {
    ldsA[bfi] = (unsigned int)(size_t)&As[bfi][0][0];
    ldsB[bfi] = (unsigned int)(size_t)&Bs[bfi][0][0];
  }
  if (wv == 0) {
    tdm_load_2d(ldsA[0], A + (size_t)mBase * E_c, KSTEP, remRows, KSTEP, 128,
                (unsigned long long)E_c, true);
    tdm_load_2d(ldsB[0], W + (size_t)nBase * E_c, KSTEP, 128, KSTEP, 128,
                (unsigned long long)E_c, true);
  }
#else
  const int lr = threadIdx.x >> 1;          // 0..127
  const int lc = (threadIdx.x & 1) << 4;    // 0 or 16
#endif

  for (int kt = 0; kt < NK; ++kt) {
#if HAS_TDM
    if (wv == 0) {
      if (kt + 1 < NK) {
        const int k1 = (kt + 1) * KSTEP;
        const int nb = (kt + 1) & 1;
        tdm_load_2d(ldsA[nb], A + (size_t)mBase * E_c + k1, KSTEP, remRows,
                    KSTEP, 128, (unsigned long long)E_c, true);
        tdm_load_2d(ldsB[nb], W + (size_t)nBase * E_c + k1, KSTEP, 128,
                    KSTEP, 128, (unsigned long long)E_c, true);
        __builtin_amdgcn_s_wait_tensorcnt((short)2);  // oldest pair complete
      } else {
        __builtin_amdgcn_s_wait_tensorcnt((short)0);
      }
    }
    __syncthreads();
    const int buf = kt & 1;
#else
    {  // manual staging fallback (also what the host pass sees)
      const int k0 = kt * KSTEP;
      const int m = mBase + lr;
      const unsigned short* srcA = A + (size_t)m * E_c + k0 + lc;
#pragma unroll
      for (int t = 0; t < 8; ++t) {
        unsigned int vl = (m < BS_c) ? *(const unsigned int*)(srcA + 2 * t) : 0u;
        *(unsigned int*)&As[0][lr][lc + 2 * t] = vl;
      }
      const unsigned short* srcB = W + (size_t)(nBase + lr) * E_c + k0 + lc;
#pragma unroll
      for (int t = 0; t < 8; ++t)
        *(unsigned int*)&Bs[0][lr][lc + 2 * t] = *(const unsigned int*)(srcB + 2 * t);
    }
    __syncthreads();
    const int buf = 0;
#endif

    v16bf bfr[2], afr[4];
#pragma unroll
    for (int j = 0; j < 2; ++j)
      bfr[j] = frag_b_nk(&Bs[buf][wn * 32 + j * 16][0], TDS, lane);
#pragma unroll
    for (int i = 0; i < 4; ++i)
      afr[i] = frag_a(&As[buf][wm * 64 + i * 16][0], TDS, lane);
#pragma unroll
    for (int i = 0; i < 4; ++i)
#pragma unroll
      for (int j = 0; j < 2; ++j) acc[i][j] = wmma_bf16(afr[i], bfr[j], acc[i][j]);
    __syncthreads();
  }

  // Epilogue. C layout: VGPR r -> row = r + (lane>=16)*8, col = lane&15
  const int colHalf = lane & 15;
  const int rowHalf = (lane >> 4) << 3;
#pragma unroll
  for (int i = 0; i < 4; ++i) {
    const int mSub = mBase + wm * 64 + i * 16 + rowHalf;
#pragma unroll
    for (int j = 0; j < 2; ++j) {
      const int col  = nBase + wn * 32 + j * 16 + colHalf;
      const float bc = bias[col];
#pragma unroll
      for (int r = 0; r < 8; ++r) {
        const int m = mSub + r;
        float y = acc[i][j][r] + bc;
        if constexpr (EPI == EPI_OUT) {
          if (m < BS_c) dstF[(size_t)m * E_c + col] = y;
        } else {
          const int b    = m / S_c;
          const int sidx = m - b * S_c;   // always < S_c for our grid
          const int h    = col / D_c;
          const int d    = col - h * D_c;
          if constexpr (EPI == EPI_ROPE) {
            const float cs = fcos[sidx * (D_c / 2) + (d >> 1)];
            const float sn = fsin[sidx * (D_c / 2) + (d >> 1)];
            const float partner = __shfl_xor(y, 1, 32);  // pair lane holds d^1
            y = (d & 1) ? (partner * sn + y * cs) : (y * cs - partner * sn);
          }
          if (m < BS_c)
            dstBf[(((size_t)b * H_c + h) * Sp_c + sidx) * Dp_c + d] = f2bf(y);
        }
      }
    }
  }
}

// One (b,h, 16-query-row tile) per block: scores -> softmax -> attn out -> P@V
__global__ __launch_bounds__(128) void attn_kernel(
    const unsigned short* __restrict__ qp,  // [B*H, Sp, Dp] bf16 (rope'd)
    const unsigned short* __restrict__ kp,
    const unsigned short* __restrict__ vp,
    float* __restrict__ attn,               // [B*H, S, S] f32
    unsigned short* __restrict__ ctx)       // [B*S, E] bf16
{
  extern __shared__ unsigned char smem[];
  float*          Pf = (float*)smem;                              // [16][Sp] f32
  unsigned short* Pb = (unsigned short*)(smem + 16 * Sp_c * 4);   // [16][Sp] bf16
  unsigned short* Vs = (unsigned short*)(smem + 16 * Sp_c * 6);   // [2][32][Dp] bf16

  const int lane = threadIdx.x & 31;
  const int wv   = threadIdx.x >> 5;   // 0..3
  const int m0   = blockIdx.x * 16;
  const int bh   = blockIdx.y;
  const float rscale = 0.10660035817780522f;  // 1/sqrt(88)
  constexpr int NKV = Sp_c / KSTEP;           // 33
  constexpr int VTILE = KSTEP * Dp_c;         // 3072 elements per V tile

  const int rowHalf = (lane >> 4) << 3;
  const int cl      = lane & 15;

  // Q fragments stay register-resident: 3 K-steps cover Dp=96
  const unsigned short* qbase = qp + ((size_t)bh * Sp_c + m0) * Dp_c;
  v16bf aq[3];
#pragma unroll
  for (int ks = 0; ks < 3; ++ks) aq[ks] = frag_a(qbase + ks * KSTEP, Dp_c, lane);

#if HAS_TDM
  // Kick off the first V-tile DMA now; it overlaps scores + softmax.
  if (wv == 0)
    tdm_load_2d((unsigned int)(size_t)Vs, vp + (size_t)bh * Sp_c * Dp_c,
                VTILE, 1, VTILE, 1, (unsigned long long)VTILE, false);
#endif

  v8f zero8 = {0.f, 0.f, 0.f, 0.f, 0.f, 0.f, 0.f, 0.f};

  // Phase 1: scores = Q K^T / sqrt(D); waves stripe over 66 column tiles
  for (int jt = wv; jt < Sp_c / 16; jt += 4) {
    const unsigned short* kbase = kp + ((size_t)bh * Sp_c + jt * 16) * Dp_c;
    v8f acc = zero8;
#pragma unroll
    for (int ks = 0; ks < 3; ++ks)
      acc = wmma_bf16(aq[ks], frag_b_nk(kbase + ks * KSTEP, Dp_c, lane), acc);
    const int col = jt * 16 + cl;
#pragma unroll
    for (int r = 0; r < 8; ++r)
      Pf[(rowHalf + r) * Sp_c + col] = (col < S_c) ? acc[r] * rscale : -3.0e38f;
  }
  __syncthreads();

  // Phase 2: softmax per row (8 lanes per row), write fp32 attn, bf16 probs
  {
    const int row = threadIdx.x >> 3;
    const int sub = threadIdx.x & 7;
    float mx = -3.0e38f;
    for (int c = sub; c < Sp_c; c += 8) mx = fmaxf(mx, Pf[row * Sp_c + c]);
    mx = fmaxf(mx, __shfl_xor(mx, 4, 8));
    mx = fmaxf(mx, __shfl_xor(mx, 2, 8));
    mx = fmaxf(mx, __shfl_xor(mx, 1, 8));
    float sum = 0.f;
    for (int c = sub; c < Sp_c; c += 8) {
      float e = __expf(Pf[row * Sp_c + c] - mx);
      Pf[row * Sp_c + c] = e;
      sum += e;
    }
    sum += __shfl_xor(sum, 4, 8);
    sum += __shfl_xor(sum, 2, 8);
    sum += __shfl_xor(sum, 1, 8);
    const float inv = 1.0f / sum;
    const int grow = m0 + row;
    float* orow = attn + ((size_t)bh * S_c + grow) * S_c;
    for (int c = sub; c < Sp_c; c += 8) {
      const float pv = Pf[row * Sp_c + c] * inv;
      Pb[row * Sp_c + c] = f2bf(pv);   // pad cols become exact 0
      if (grow < S_c && c < S_c) orow[c] = pv;
    }
  }
  __syncthreads();

  // Phase 3: ctx = P @ V (16 x Dp), waves stripe over 6 N-subtiles
  v8f accv[2];
  accv[0] = zero8; accv[1] = zero8;
  for (int kt = 0; kt < NKV; ++kt) {
#if HAS_TDM
    if (wv == 0) {
      if (kt + 1 < NKV) {
        tdm_load_2d((unsigned int)(size_t)(Vs + ((kt + 1) & 1) * VTILE),
                    vp + ((size_t)bh * Sp_c + (kt + 1) * KSTEP) * Dp_c,
                    VTILE, 1, VTILE, 1, (unsigned long long)VTILE, false);
        __builtin_amdgcn_s_wait_tensorcnt((short)1);  // oldest tile complete
      } else {
        __builtin_amdgcn_s_wait_tensorcnt((short)0);
      }
    }
    __syncthreads();
    unsigned short* vtile = Vs + (kt & 1) * VTILE;
#else
    {  // manual staging fallback
      const int jr = threadIdx.x >> 2;
      const int q4 = (threadIdx.x & 3) * 24;
      const uint4* s =
          (const uint4*)(vp + ((size_t)bh * Sp_c + kt * KSTEP + jr) * Dp_c + q4);
      uint4 d0 = s[0], d1 = s[1], d2 = s[2];
      uint4* dst = (uint4*)(Vs + jr * Dp_c + q4);
      dst[0] = d0; dst[1] = d1; dst[2] = d2;
    }
    __syncthreads();
    unsigned short* vtile = Vs;
#endif
    v16bf ap = frag_a(Pb + kt * KSTEP, Sp_c, lane);
#pragma unroll
    for (int u = 0; u < 2; ++u) {
      const int nt = wv + u * 4;           // wave-uniform
      if (nt < 6) {
        v16bf bv = frag_b_kn(vtile + nt * 16, Dp_c, lane);
        accv[u] = wmma_bf16(ap, bv, accv[u]);
      }
    }
    __syncthreads();
  }

  const int b = bh >> 4, h = bh & 15;
#pragma unroll
  for (int u = 0; u < 2; ++u) {
    const int nt = wv + u * 4;
    if (nt >= 6) continue;
    const int d = nt * 16 + cl;
#pragma unroll
    for (int r = 0; r < 8; ++r) {
      const int grow = m0 + rowHalf + r;
      if (d < D_c && grow < S_c)
        ctx[((size_t)b * S_c + grow) * E_c + h * D_c + d] = f2bf(accv[u][r]);
    }
  }
}

extern "C" void kernel_launch(void* const* d_in, const int* in_sizes, int n_in,
                              void* d_out, int out_size, void* d_ws, size_t ws_size,
                              hipStream_t stream) {
  (void)in_sizes; (void)n_in; (void)out_size; (void)ws_size;

  const float* hs   = (const float*)d_in[0];
  const float* fcos = (const float*)d_in[1];
  const float* fsin = (const float*)d_in[2];
  const float* q_w  = (const float*)d_in[3];
  const float* q_b  = (const float*)d_in[4];
  const float* k_w  = (const float*)d_in[5];
  const float* k_b  = (const float*)d_in[6];
  const float* v_w  = (const float*)d_in[7];
  const float* v_b  = (const float*)d_in[8];
  const float* o_w  = (const float*)d_in[9];
  const float* o_b  = (const float*)d_in[10];

  float* out  = (float*)d_out;                       // [B,S,E] f32
  float* attn = out + (size_t)BS_c * E_c;            // [B,H,S,S] f32

  unsigned char* ws = (unsigned char*)d_ws;
  size_t off = 0;
  auto carve = [&](size_t bytes) -> void* {
    void* p = ws + off;
    off = (off + bytes + 255) & ~(size_t)255;
    return p;
  };
  const size_t wBytes   = (size_t)E_c * E_c * 2;
  const size_t xBytes   = (size_t)BS_c * E_c * 2;
  const size_t padBytes = (size_t)B_c * H_c * Sp_c * Dp_c * 2;

  unsigned short* wq  = (unsigned short*)carve(wBytes);
  unsigned short* wk  = (unsigned short*)carve(wBytes);
  unsigned short* wvv = (unsigned short*)carve(wBytes);
  unsigned short* wo  = (unsigned short*)carve(wBytes);
  unsigned short* xb  = (unsigned short*)carve(xBytes);
  unsigned short* qp  = (unsigned short*)carve(padBytes);
  unsigned short* kp  = (unsigned short*)carve(padBytes);
  unsigned short* vp  = (unsigned short*)carve(padBytes);
  unsigned short* ctx = (unsigned short*)carve(xBytes);

  // 1) precision conversion (bf16 feeds the WMMA pipes)
  cvt_f32_bf16<<<1024, 256, 0, stream>>>(q_w, wq,  (long)E_c * E_c);
  cvt_f32_bf16<<<1024, 256, 0, stream>>>(k_w, wk,  (long)E_c * E_c);
  cvt_f32_bf16<<<1024, 256, 0, stream>>>(v_w, wvv, (long)E_c * E_c);
  cvt_f32_bf16<<<1024, 256, 0, stream>>>(o_w, wo,  (long)E_c * E_c);
  cvt_f32_bf16<<<2048, 256, 0, stream>>>(hs,  xb,  (long)BS_c * E_c);

  // 2) zero padded q/k/v so pad rows/cols contribute exact zeros to WMMA
  const long padElems = (long)B_c * H_c * Sp_c * Dp_c;
  zero_u16<<<2048, 256, 0, stream>>>(qp, padElems);
  zero_u16<<<2048, 256, 0, stream>>>(kp, padElems);
  zero_u16<<<2048, 256, 0, stream>>>(vp, padElems);

  // 3) QKV projections with fused bias + RoPE + transpose-to-heads
  dim3 gg(E_c / 128, (BS_c + 127) / 128);   // (11, 65)
  gemm_kernel<EPI_ROPE><<<gg, 256, 0, stream>>>(xb, wq,  q_b, fcos, fsin, qp, nullptr);
  gemm_kernel<EPI_ROPE><<<gg, 256, 0, stream>>>(xb, wk,  k_b, fcos, fsin, kp, nullptr);
  gemm_kernel<EPI_V   ><<<gg, 256, 0, stream>>>(xb, wvv, v_b, nullptr, nullptr, vp, nullptr);

  // 4) fused scores + softmax + attn write + P@V (rows resident in big LDS)
  const size_t smemBytes =
      (size_t)16 * Sp_c * 6 + (size_t)2 * KSTEP * Dp_c * 2;  // ~113.6 KB
  hipFuncSetAttribute((const void*)attn_kernel,
                      hipFuncAttributeMaxDynamicSharedMemorySize, (int)smemBytes);
  attn_kernel<<<dim3((S_c + 15) / 16, B_c * H_c), 128, smemBytes, stream>>>(
      qp, kp, vp, attn, ctx);

  // 5) output projection
  gemm_kernel<EPI_OUT><<<gg, 256, 0, stream>>>(ctx, wo, o_b, nullptr, nullptr,
                                               nullptr, out);
}
